// SeqAttnContextSecondHop_20023137534647
// MI455X (gfx1250) — matrix operand order repaired and checked by reading, not compiled, not run on hardware
//
#include <hip/hip_runtime.h>
#include <hip/hip_bf16.h>
#include <stdint.h>

// ---------------------------------------------------------------------------
// Problem constants (from reference): B=16, LX=LY=2048, XD=YD=OD=1024
// ---------------------------------------------------------------------------
#define BATCH 16
#define SEQ   2048
#define DIM   1024

typedef __bf16 v16bf __attribute__((ext_vector_type(16)));
typedef __bf16 v8bf  __attribute__((ext_vector_type(8)));
typedef float  v8f   __attribute__((ext_vector_type(8)));

// ---------------------------------------------------------------------------
// CDNA5 async global->LDS copy (GLOBAL_LOAD_ASYNC_TO_LDS_B128, ASYNCcnt)
// Builtin prototype (from hipcc diagnostic): arg0 = int4 addrspace(1)*,
// arg1 = int4 addrspace(3)*, then two immediates (offset, cpol).
// ---------------------------------------------------------------------------
#if __has_builtin(__builtin_amdgcn_global_load_async_to_lds_b128)
#define HAVE_ASYNC_LDS 1
typedef int v4i __attribute__((vector_size(16)));
typedef v4i __attribute__((address_space(1)))* gv4i_ptr;
typedef v4i __attribute__((address_space(3)))* lv4i_ptr;
static __device__ __forceinline__ void async_cp16(void* lds, const void* g) {
    v4i* gg = (v4i*)g;      // reinterpret (and drop const) in generic AS
    v4i* ll = (v4i*)lds;
    __builtin_amdgcn_global_load_async_to_lds_b128((gv4i_ptr)gg, (lv4i_ptr)ll, 0, 0);
}
static __device__ __forceinline__ void async_wait0() {
    asm volatile("s_wait_asynccnt 0x0" ::: "memory");
}
#else
#define HAVE_ASYNC_LDS 0
#endif

// ---------------------------------------------------------------------------
// helpers
// ---------------------------------------------------------------------------
static __device__ __forceinline__ unsigned short bf_bits(float f) {
    unsigned u = __builtin_bit_cast(unsigned, f);
    u += 0x7fffu + ((u >> 16) & 1u);          // round-to-nearest-even
    return (unsigned short)(u >> 16);
}
static __device__ __forceinline__ __bf16 f2bf(float f) {
    unsigned short s = bf_bits(f);
    return __builtin_bit_cast(__bf16, s);
}
static __device__ __forceinline__ unsigned pack2(float a, float b) {
    return (unsigned)bf_bits(a) | ((unsigned)bf_bits(b) << 16);
}
static __device__ __forceinline__ v16bf mk16(v8bf lo, v8bf hi) {
    return __builtin_shufflevector(lo, hi, 0,1,2,3,4,5,6,7,8,9,10,11,12,13,14,15);
}
static __device__ __forceinline__ v8f vzero8() {
    v8f z;
#pragma unroll
    for (int i = 0; i < 8; ++i) z[i] = 0.0f;
    return z;
}
static __device__ __forceinline__ v8f wmma_bf16(v16bf a, v16bf b, v8f c) {
    // v_wmma_f32_16x16x32_bf16  (neg_a, A, neg_b, B, c_mod, C, reuse_a, reuse_b)
    return __builtin_amdgcn_wmma_f32_16x16x32_bf16(false, a, false, b, (short)0, c,
                                                   false, false);
}
static __device__ __forceinline__ float red_max16(float v) {
#pragma unroll
    for (int m = 1; m < 16; m <<= 1) v = fmaxf(v, __shfl_xor(v, m, 32));
    return v;
}
static __device__ __forceinline__ float red_sum16(float v) {
#pragma unroll
    for (int m = 1; m < 16; m <<= 1) v += __shfl_xor(v, m, 32);
    return v;
}

// ---------------------------------------------------------------------------
// Kernel 1: convert W (D x OD fp32, row-major over k) -> Wt (OD x D bf16)
// so GEMM B-fragments read contiguous k-runs from a Wt row.
// ---------------------------------------------------------------------------
__global__ __launch_bounds__(256) void wt_kernel(const float* __restrict__ Wq,
                                                 const float* __restrict__ Wk,
                                                 const float* __restrict__ Wv,
                                                 __bf16* __restrict__ wt) {
    __shared__ float tile[32][33];
    const float* src = (blockIdx.z == 0) ? Wq : (blockIdx.z == 1) ? Wk : Wv;
    __bf16* dst = wt + (size_t)blockIdx.z * DIM * DIM;
    int tx = threadIdx.x & 31, ty = threadIdx.x >> 5;   // 32 x 8
    int kb = blockIdx.x * 32, nb = blockIdx.y * 32;
#pragma unroll
    for (int i = 0; i < 32; i += 8)
        tile[ty + i][tx] = src[(size_t)(kb + ty + i) * DIM + nb + tx];
    __syncthreads();
#pragma unroll
    for (int i = 0; i < 32; i += 8)
        dst[(size_t)(nb + ty + i) * DIM + kb + tx] = f2bf(tile[tx][ty + i]);
}

// ---------------------------------------------------------------------------
// Kernel 2: P = relu(X @ W + bias), X fp32 (R x 1024), Wt bf16 (1024 x 1024),
// output bf16 either row-major (outN) or transposed per-batch (outT: B,OD,SEQ).
// 256 threads = 8 waves; block tile 128 x 128; wave = 16 rows x 128 cols.
// X global loads are software-pipelined one k-step ahead; Wt tiles are staged
// with async global->LDS copies when available.
// ---------------------------------------------------------------------------
#define XS_STRIDE 40   // 32 + 8 pad (bf16 elems)
#define WS_STRIDE 40

__global__ __launch_bounds__(256) void proj_kernel(const float* __restrict__ X,
                                                   const __bf16* __restrict__ Wt,
                                                   const float* __restrict__ bias,
                                                   __bf16* __restrict__ outN,
                                                   __bf16* __restrict__ outT) {
    __shared__ __bf16 Xs[128 * XS_STRIDE];
    __shared__ __bf16 Wsh[128 * WS_STRIDE];

    const int tid  = threadIdx.x;
    const int w    = tid >> 5;
    const int lane = tid & 31;
    const int half = lane >> 4;
    const int l16  = lane & 15;

    const int rowBase = blockIdx.x * 128;
    const int colBase = blockIdx.y * 128;

    v8f acc[8];
#pragma unroll
    for (int c = 0; c < 8; ++c) acc[c] = vzero8();

    // software pipeline: X tile (fp32, needs VALU convert) prefetched to regs
    float4 xv[4];
#pragma unroll
    for (int i = 0; i < 4; ++i) {
        int chunk = tid + 256 * i;                // 1024 chunks of 4 floats
        int r = chunk >> 3, c4 = chunk & 7;
        xv[i] = *(const float4*)&X[(size_t)(rowBase + r) * DIM + c4 * 4];
    }

    for (int kk = 0; kk < DIM / 32; ++kk) {
        const int k0 = kk * 32;
        __syncthreads();                          // previous tile consumed

        // stage Wt tile (128 x 32 bf16): async DMA straight into LDS
#pragma unroll
        for (int i = 0; i < 2; ++i) {
            int chunk = tid + 256 * i;            // 512 chunks of 8 bf16
            int r = chunk >> 2, c8 = chunk & 3;
            const __bf16* src = &Wt[(size_t)(colBase + r) * DIM + k0 + c8 * 8];
#if HAVE_ASYNC_LDS
            async_cp16(&Wsh[r * WS_STRIDE + c8 * 8], src);
#else
            *(uint4*)&Wsh[r * WS_STRIDE + c8 * 8] = *(const uint4*)src;
#endif
        }
        // convert + store prefetched X tile
#pragma unroll
        for (int i = 0; i < 4; ++i) {
            int chunk = tid + 256 * i;
            int r = chunk >> 3, c4 = chunk & 7;
            uint2 p;
            p.x = pack2(xv[i].x, xv[i].y);
            p.y = pack2(xv[i].z, xv[i].w);
            *(uint2*)&Xs[r * XS_STRIDE + c4 * 4] = p;
        }
        // prefetch next X tile while WMMAs run
        if (kk + 1 < DIM / 32) {
#pragma unroll
            for (int i = 0; i < 4; ++i) {
                int chunk = tid + 256 * i;
                int r = chunk >> 3, c4 = chunk & 7;
                xv[i] = *(const float4*)&X[(size_t)(rowBase + r) * DIM + k0 + 32 + c4 * 4];
            }
        }
#if HAVE_ASYNC_LDS
        async_wait0();                            // this wave's async writes landed
#endif
        __syncthreads();

        const __bf16* arow = &Xs[(w * 16 + l16) * XS_STRIDE];
        const int a0 = half * 8;
        v16bf afrag = mk16(*(const v8bf*)(arow + a0), *(const v8bf*)(arow + a0 + 16));
#pragma unroll
        for (int c = 0; c < 8; ++c) {
            const __bf16* brow = &Wsh[(c * 16 + l16) * WS_STRIDE + half * 8];
            v16bf bfrag = mk16(*(const v8bf*)brow, *(const v8bf*)(brow + 16));
            acc[c] = wmma_bf16(afrag, bfrag, acc[c]);
        }
    }

    if (outT) {  // V path: store transposed (B, OD, SEQ)
        const int b      = rowBase / SEQ;
        const int lyBase = (rowBase % SEQ) + w * 16 + half * 8;
#pragma unroll
        for (int c = 0; c < 8; ++c) {
            int od = colBase + c * 16 + l16;
            float bi = bias[od];
            v8bf pk;
#pragma unroll
            for (int j = 0; j < 8; ++j)
                pk[j] = f2bf(fmaxf(acc[c][j] + bi, 0.0f));
            *(v8bf*)&outT[(size_t)b * DIM * SEQ + (size_t)od * SEQ + lyBase] = pk;
        }
    } else {     // Q/K path: row-major
#pragma unroll
        for (int c = 0; c < 8; ++c) {
            int od = colBase + c * 16 + l16;
            float bi = bias[od];
#pragma unroll
            for (int j = 0; j < 8; ++j) {
                int row = rowBase + w * 16 + half * 8 + j;
                outN[(size_t)row * DIM + od] = f2bf(fmaxf(acc[c][j] + bi, 0.0f));
            }
        }
    }
}

// ---------------------------------------------------------------------------
// Kernel 3: softmax stats pass.  Each wave owns 16 query rows; Q fragments
// live in 256 VGPRs; loop over 2048 keys in 16-key column tiles.
// ---------------------------------------------------------------------------
__global__ __launch_bounds__(256) void stats_kernel(const __bf16* __restrict__ q,
                                                    const __bf16* __restrict__ k,
                                                    const unsigned char* __restrict__ mask,
                                                    float* __restrict__ mOut,
                                                    float* __restrict__ lOut) {
    const int w    = threadIdx.x >> 5;
    const int lane = threadIdx.x & 31;
    const int half = lane >> 4;
    const int l16  = lane & 15;
    const int qRow0 = blockIdx.x * 128 + w * 16;    // global query row base
    const int b = qRow0 >> 11;                       // /2048

    // load this wave's 16x1024 Q tile as 32 A-fragments (kept in VGPRs)
    v16bf qf[32];
    const __bf16* qrow = q + (size_t)(qRow0 + l16) * DIM;
#pragma unroll
    for (int kf = 0; kf < 32; ++kf) {
        int base = kf * 32 + half * 8;
        qf[kf] = mk16(*(const v8bf*)(qrow + base), *(const v8bf*)(qrow + base + 16));
    }

    v8f m, l;
#pragma unroll
    for (int j = 0; j < 8; ++j) { m[j] = -__builtin_inff(); l[j] = 0.0f; }

    const __bf16* kbatch = k + (size_t)b * SEQ * DIM;
    const unsigned char* mb = mask + b * SEQ;

    for (int kt = 0; kt < SEQ / 16; ++kt) {
        const int key = kt * 16 + l16;
        const __bf16* krow = kbatch + (size_t)key * DIM;
        __builtin_prefetch(krow + 16 * DIM, 0, 1);   // global_prefetch_b8
        v8f s = vzero8();
#pragma unroll
        for (int kf = 0; kf < 32; ++kf) {
            int base = kf * 32 + half * 8;
            v16bf bb = mk16(*(const v8bf*)(krow + base), *(const v8bf*)(krow + base + 16));
            s = wmma_bf16(qf[kf], bb, s);
        }
        const bool masked = (mb[kt * 16 + l16] != 0);
#pragma unroll
        for (int j = 0; j < 8; ++j) {
            float sj   = masked ? -__builtin_inff() : s[j];
            float rm   = red_max16(sj);
            float mnew = fmaxf(m[j], rm);
            float alpha = (mnew > -3.3e38f) ? __expf(m[j] - mnew) : 0.0f;
            float p     = masked ? 0.0f : __expf(s[j] - mnew);
            float rs    = red_sum16(p);
            l[j] = l[j] * alpha + rs;
            m[j] = mnew;
        }
    }
    if (l16 == 0) {
#pragma unroll
        for (int j = 0; j < 8; ++j) {
            int row = qRow0 + half * 8 + j;
            mOut[row] = m[j];
            lOut[row] = l[j];
        }
    }
}

// ---------------------------------------------------------------------------
// Kernel 4: context pass.  Block = 32 query rows; Q tile in LDS (64.5 KB of
// the 320 KB WGP LDS, async-DMA'd).  Per 128-key block: each wave computes a
// 16-key S column tile, writes normalized P (bf16) to LDS, then all waves
// multiply P @ V on disjoint 128-wide OD slices (Vt pre-transposed).
// ---------------------------------------------------------------------------
#define QS_STRIDE 1032   // 1024 + 8 pad
#define PS_STRIDE 136    // 128 + 8 pad

__global__ __launch_bounds__(256) void context_kernel(const __bf16* __restrict__ q,
                                                      const __bf16* __restrict__ k,
                                                      const __bf16* __restrict__ vt,
                                                      const unsigned char* __restrict__ mask,
                                                      const float* __restrict__ mIn,
                                                      const float* __restrict__ lIn,
                                                      float* __restrict__ out) {
    __shared__ __bf16 Qs[32 * QS_STRIDE];
    __shared__ __bf16 Ps[32 * PS_STRIDE];

    const int tid  = threadIdx.x;
    const int w    = tid >> 5;
    const int lane = tid & 31;
    const int half = lane >> 4;
    const int l16  = lane & 15;

    const int qBase = blockIdx.x * 32;               // global query row base
    const int b = qBase >> 11;

    // cooperative Q tile load (32 x 1024 bf16), async DMA when available
#pragma unroll
    for (int i = 0; i < 8; ++i) {
        int chunk = tid + 256 * i;                   // 2048 chunks of 8 bf16
        int r = chunk >> 6, c = chunk & 63;
        const __bf16* src = &q[(size_t)(qBase + r) * DIM + c * 8];
#if HAVE_ASYNC_LDS
        async_cp16(&Qs[r * QS_STRIDE + c * 8], src);
#else
        *(uint4*)&Qs[r * QS_STRIDE + c * 8] = *(const uint4*)src;
#endif
    }

    // per-lane softmax stats in the WMMA C layout
    float mreg[2][8], ilreg[2][8];
#pragma unroll
    for (int rt = 0; rt < 2; ++rt)
#pragma unroll
        for (int j = 0; j < 8; ++j) {
            int row = qBase + rt * 16 + half * 8 + j;
            mreg[rt][j]  = mIn[row];
            ilreg[rt][j] = 1.0f / lIn[row];
        }
#if HAVE_ASYNC_LDS
    async_wait0();
#endif
    __syncthreads();

    v8f acc[2][8];
#pragma unroll
    for (int rt = 0; rt < 2; ++rt)
#pragma unroll
        for (int ct = 0; ct < 8; ++ct) acc[rt][ct] = vzero8();

    const __bf16* kb = k  + (size_t)b * SEQ * DIM;
    const __bf16* vb = vt + (size_t)b * DIM * SEQ;
    const unsigned char* mbm = mask + b * SEQ;

    for (int blk = 0; blk < SEQ / 128; ++blk) {
        const int keyBase = blk * 128;
        // ---- S phase: this wave's 16 keys ----
        const int myKey = keyBase + w * 16;
        const __bf16* krow = kb + (size_t)(myKey + l16) * DIM;
        __builtin_prefetch(krow + 128 * DIM, 0, 1);
        v8f s0 = vzero8(), s1 = vzero8();
#pragma unroll
        for (int kf = 0; kf < 32; ++kf) {
            int ab = kf * 32 + half * 8;
            v16bf a0 = mk16(*(const v8bf*)&Qs[l16 * QS_STRIDE + ab],
                            *(const v8bf*)&Qs[l16 * QS_STRIDE + ab + 16]);
            v16bf a1 = mk16(*(const v8bf*)&Qs[(16 + l16) * QS_STRIDE + ab],
                            *(const v8bf*)&Qs[(16 + l16) * QS_STRIDE + ab + 16]);
            int bb = kf * 32 + half * 8;
            v16bf kfrag = mk16(*(const v8bf*)(krow + bb), *(const v8bf*)(krow + bb + 16));
            s0 = wmma_bf16(a0, kfrag, s0);
            s1 = wmma_bf16(a1, kfrag, s1);
        }
        const bool masked = (mbm[myKey + l16] != 0);
#pragma unroll
        for (int rt = 0; rt < 2; ++rt) {
            v8f s = rt ? s1 : s0;
#pragma unroll
            for (int j = 0; j < 8; ++j) {
                float p = masked ? 0.0f
                                 : __expf(s[j] - mreg[rt][j]) * ilreg[rt][j];
                Ps[(rt * 16 + half * 8 + j) * PS_STRIDE + w * 16 + l16] = f2bf(p);
            }
        }
        __syncthreads();
        // ---- PV phase: wave handles OD slice [w*128, w*128+128) ----
#pragma unroll
        for (int kf = 0; kf < 4; ++kf) {
            int ab = kf * 32 + half * 8;
            v16bf a0 = mk16(*(const v8bf*)&Ps[l16 * PS_STRIDE + ab],
                            *(const v8bf*)&Ps[l16 * PS_STRIDE + ab + 16]);
            v16bf a1 = mk16(*(const v8bf*)&Ps[(16 + l16) * PS_STRIDE + ab],
                            *(const v8bf*)&Ps[(16 + l16) * PS_STRIDE + ab + 16]);
            int koff = keyBase + kf * 32 + half * 8;
#pragma unroll
            for (int ct = 0; ct < 8; ++ct) {
                int od = w * 128 + ct * 16 + l16;
                const __bf16* vrow = vb + (size_t)od * SEQ + koff;
                v16bf vfrag = mk16(*(const v8bf*)vrow, *(const v8bf*)(vrow + 16));
                acc[0][ct] = wmma_bf16(a0, vfrag, acc[0][ct]);
                acc[1][ct] = wmma_bf16(a1, vfrag, acc[1][ct]);
            }
        }
        __syncthreads();
    }

    // store fp32 contexts
#pragma unroll
    for (int rt = 0; rt < 2; ++rt)
#pragma unroll
        for (int ct = 0; ct < 8; ++ct) {
            int od = w * 128 + ct * 16 + l16;
#pragma unroll
            for (int j = 0; j < 8; ++j) {
                int row = qBase + rt * 16 + half * 8 + j;
                out[(size_t)row * DIM + od] = acc[rt][ct][j];
            }
        }
}

// ---------------------------------------------------------------------------
// launch
// ---------------------------------------------------------------------------
extern "C" void kernel_launch(void* const* d_in, const int* in_sizes, int n_in,
                              void* d_out, int out_size, void* d_ws, size_t ws_size,
                              hipStream_t stream) {
    const float* x  = (const float*)d_in[0];
    const float* y  = (const float*)d_in[1];
    const unsigned char* y_mask = (const unsigned char*)d_in[2];
    const float* Wq = (const float*)d_in[3];
    const float* bq = (const float*)d_in[4];
    const float* Wk = (const float*)d_in[5];
    const float* bk = (const float*)d_in[6];
    const float* Wv = (const float*)d_in[7];
    const float* bv = (const float*)d_in[8];
    float* out = (float*)d_out;

    char* ws = (char*)d_ws;
    const size_t rows  = (size_t)BATCH * SEQ;           // 32768
    const size_t nProj = rows * DIM * sizeof(__bf16);   // 64 MB
    __bf16* qb  = (__bf16*)(ws);
    __bf16* kb  = (__bf16*)(ws + nProj);
    __bf16* vtb = (__bf16*)(ws + 2 * nProj);
    __bf16* wt  = (__bf16*)(ws + 3 * nProj);
    float*  mb  = (float*)(ws + 3 * nProj + 3 * (size_t)DIM * DIM * sizeof(__bf16));
    float*  lb  = mb + rows;

    // 1) weight convert + transpose (fp32 -> bf16, N-major)
    wt_kernel<<<dim3(32, 32, 3), 256, 0, stream>>>(Wq, Wk, Wv, wt);

    // 2) projections (relu(XW+b)); V stored transposed
    dim3 pgrid(rows / 128, DIM / 128);
    proj_kernel<<<pgrid, 256, 0, stream>>>(x, wt + 0 * DIM * DIM, bq, qb, nullptr);
    proj_kernel<<<pgrid, 256, 0, stream>>>(y, wt + 1 * DIM * DIM, bk, kb, nullptr);
    proj_kernel<<<pgrid, 256, 0, stream>>>(y, wt + 2 * DIM * DIM, bv, nullptr, vtb);

    // 3) softmax stats (m, l) per query row
    stats_kernel<<<rows / 128, 256, 0, stream>>>(qb, kb, y_mask, mb, lb);

    // 4) contexts
    context_kernel<<<rows / 32, 256, 0, stream>>>(qb, kb, vtb, y_mask, mb, lb, out);

    (void)in_sizes; (void)n_in; (void)out_size; (void)ws_size;
}
